// EncoderBlock_Mamba_72335839199747
// MI455X (gfx1250) — matrix operand order repaired and checked by reading, not compiled, not run on hardware
//
#include <hip/hip_runtime.h>
#include <hip/hip_bf16.h>

// ---------------------------------------------------------------------------
// Problem constants (match reference)
// ---------------------------------------------------------------------------
#define B_   4
#define C_   64
#define H_   128
#define W_   128
#define L_   (H_ * W_)          // 16384
#define NT_  (B_ * L_)          // 65536 tokens
#define DI_  128                // D_INNER
#define DS_  16                 // D_STATE
#define DR_  4                  // DT_RANK
#define OC_  128                // OUT_C
#define DBL_STRIDE 48           // padded 36 -> 48 (3 WMMA n-tiles)
#define NCH_ 64                 // scan chunks per sequence
#define CH_  (L_ / NCH_)        // 256 steps per chunk
#define EPS_ 1e-5f
#define POOLED_ELEMS (B_ * OC_ * (H_/2) * (W_/2))   // 2,097,152
#define SKIP_ELEMS   (B_ * C_ * H_ * W_)            // 4,194,304

typedef _Float16 v16h __attribute__((ext_vector_type(16)));
typedef float    v8f  __attribute__((ext_vector_type(8)));

// ---------------------------------------------------------------------------
// WMMA helpers (gfx1250: D = A(16x32 f16) * B(32x16 f16) + C(16x16 f32))
// ---------------------------------------------------------------------------
__device__ __forceinline__ v8f wmma_f16(v16h a, v16h b, v8f c) {
    return __builtin_amdgcn_wmma_f32_16x16x32_f16(
        /*neg_a=*/false, a, /*neg_b=*/false, b,
        /*c_mod=*/(short)0, c, /*reuse_a=*/false, /*reuse_b=*/false);
}

// A fragment: 16x32 tile, per the ISA VGPR layout.
// lanes 0-15: M=lane, K = k0+{0..7} and k0+16+{0..7}
// lanes16-31: M=lane-16, K = k0+8+{0..7} and k0+24+{0..7}
__device__ __forceinline__ v16h load_a_frag(const _Float16* tile_row0,
                                            int row_stride, int k0, int lane) {
    int m  = lane & 15;
    int kb = k0 + ((lane < 16) ? 0 : 8);
    const _Float16* p = tile_row0 + (size_t)m * row_stride + kb;
    v16h a;
#pragma unroll
    for (int e = 0; e < 8; ++e) { a[e] = p[e]; a[8 + e] = p[16 + e]; }
    return a;
}

// B fragment from f32 row-major weights W[N][K] (B = W^T, 32x16 tile).
// lanes 0-15: N=n0+lane, K=k0+{0..7},k0+16+{0..7}; lanes16-31: N=n0+lane-16, K=k0+8+{0..7},k0+24+{0..7}
__device__ __forceinline__ v16h load_b_frag_f32(const float* W, int ldw,
                                                int n0, int k0, int nmax, int lane) {
    int n  = n0 + (lane & 15);
    int kb = k0 + ((lane < 16) ? 0 : 8);
    v16h b;
    if (n < nmax) {
        const float* p = W + (size_t)n * ldw + kb;
#pragma unroll
        for (int e = 0; e < 8; ++e) { b[e] = (_Float16)p[e]; b[8 + e] = (_Float16)p[16 + e]; }
    } else {
#pragma unroll
        for (int e = 0; e < 16; ++e) b[e] = (_Float16)0.0f;
    }
    return b;
}

__device__ __forceinline__ float sigmoidf_(float x) { return 1.0f / (1.0f + __expf(-x)); }
__device__ __forceinline__ float siluf_(float x)    { return x * sigmoidf_(x); }

// ===========================================================================
// K1: LayerNorm(64ch) + GEMM xz = tn @ W_in^T  (M=NT, K=64, N=256)
//     writes xm_raw (NT,128) and z (NT,128)
// ===========================================================================
__global__ void k_ln_gemm_in(const float* __restrict__ x,
                             const float* __restrict__ ln_w,
                             const float* __restrict__ ln_b,
                             const float* __restrict__ W_in,
                             float* __restrict__ xm_raw,
                             float* __restrict__ z) {
    __shared__ float    xs[C_][132];      // x tile [c][l]
    __shared__ _Float16 ah[128][72];      // normalized A tile [token][c], f16

    const int tid   = threadIdx.x;
    const int blk   = blockIdx.x;         // 512 blocks
    const int b     = blk >> 7;
    const int lbase = (blk & 127) * 128;
    const float* xb = x + (size_t)b * C_ * L_;

    for (int i = tid; i < C_ * 128; i += 256) {
        int c = i >> 7, l = i & 127;
        xs[c][l] = xb[(size_t)c * L_ + lbase + l];
    }
    __syncthreads();

    if (tid < 128) {
        float s = 0.f, sq = 0.f;
#pragma unroll
        for (int c = 0; c < C_; ++c) { float v = xs[c][tid]; s += v; sq += v * v; }
        float mu  = s * (1.0f / C_);
        float var = sq * (1.0f / C_) - mu * mu;
        float inv = rsqrtf(var + EPS_);
#pragma unroll
        for (int c = 0; c < C_; ++c)
            ah[tid][c] = (_Float16)((xs[c][tid] - mu) * inv * ln_w[c] + ln_b[c]);
    }
    __syncthreads();

    const int wv = tid >> 5, lane = tid & 31;
    const int n0 = wv * 32;                       // 2 n-tiles per wave
    v16h b00 = load_b_frag_f32(W_in, 64, n0,      0, 256, lane);
    v16h b01 = load_b_frag_f32(W_in, 64, n0,     32, 256, lane);
    v16h b10 = load_b_frag_f32(W_in, 64, n0 + 16, 0, 256, lane);
    v16h b11 = load_b_frag_f32(W_in, 64, n0 + 16, 32, 256, lane);

    const int row0 = (lane < 16) ? 0 : 8;
    const int col  = lane & 15;
    const size_t tbase = (size_t)b * L_ + lbase;
    float* outbuf = (wv < 4) ? xm_raw : z;
    const int fo0 = (n0 & 127) + col;

    for (int mt = 0; mt < 8; ++mt) {
        v16h a0 = load_a_frag(&ah[mt * 16][0], 72, 0,  lane);
        v16h a1 = load_a_frag(&ah[mt * 16][0], 72, 32, lane);
        v8f acc0 = {}; v8f acc1 = {};
        acc0 = wmma_f16(a0, b00, acc0); acc0 = wmma_f16(a1, b01, acc0);
        acc1 = wmma_f16(a0, b10, acc1); acc1 = wmma_f16(a1, b11, acc1);
#pragma unroll
        for (int r = 0; r < 8; ++r) {
            size_t tok = tbase + mt * 16 + row0 + r;
            outbuf[(tok << 7) + fo0]      = acc0[r];
            outbuf[(tok << 7) + fo0 + 16] = acc1[r];
        }
    }
}

// ===========================================================================
// K2: causal depthwise conv1d (4 taps along l) + bias + SiLU
// ===========================================================================
__global__ void k_conv_silu(const float* __restrict__ xm_raw,
                            const float* __restrict__ conv_w,
                            const float* __restrict__ conv_b,
                            float* __restrict__ xm_act,
                            _Float16* __restrict__ xm_act_h) {
    size_t t = (size_t)blockIdx.x * 256 + threadIdx.x;   // NT*128
    int d = (int)(t & 127);
    size_t tok = t >> 7;
    int l = (int)(tok & (L_ - 1));
    float acc = conv_b[d];
#pragma unroll
    for (int j = 0; j < 4; ++j) {
        int off = j - 3;                 // taps at l-3..l
        if (l + off >= 0)
            acc = fmaf(conv_w[d * 4 + j], xm_raw[((tok + off) << 7) + d], acc);
    }
    float s = siluf_(acc);
    xm_act[t]   = s;
    xm_act_h[t] = (_Float16)s;
}

// ===========================================================================
// K3: dbl = xm @ W_xproj^T   (M=NT, K=128, N=36 padded to 48)
// ===========================================================================
__global__ void k_gemm_xproj(const _Float16* __restrict__ xm_act_h,
                             const float* __restrict__ W_xproj,
                             float* __restrict__ dbl) {
    const int tid = threadIdx.x, wv = tid >> 5, lane = tid & 31;
    const size_t tbase = (size_t)blockIdx.x * 128;        // 512 blocks

    v16h bf[3][4];
#pragma unroll
    for (int nt = 0; nt < 3; ++nt)
#pragma unroll
        for (int ks = 0; ks < 4; ++ks)
            bf[nt][ks] = load_b_frag_f32(W_xproj, 128, nt * 16, ks * 32, 36, lane);

    v8f acc[3] = {{}, {}, {}};
    const _Float16* arow = xm_act_h + ((tbase + wv * 16) << 7);
#pragma unroll
    for (int ks = 0; ks < 4; ++ks) {
        v16h a = load_a_frag(arow, 128, ks * 32, lane);
#pragma unroll
        for (int nt = 0; nt < 3; ++nt) acc[nt] = wmma_f16(a, bf[nt][ks], acc[nt]);
    }
    const int row0 = (lane < 16) ? 0 : 8;
    const int col  = lane & 15;
#pragma unroll
    for (int nt = 0; nt < 3; ++nt)
#pragma unroll
        for (int r = 0; r < 8; ++r) {
            size_t tok = tbase + wv * 16 + row0 + r;
            dbl[tok * DBL_STRIDE + nt * 16 + col] = acc[nt][r];
        }
}

// ===========================================================================
// K4: delta = softplus(dt @ W_dt^T + b_dt)   (K=4, pure VALU)
// ===========================================================================
__global__ void k_delta(const float* __restrict__ dbl,
                        const float* __restrict__ W_dt,
                        const float* __restrict__ b_dt,
                        float* __restrict__ delta) {
    size_t t = (size_t)blockIdx.x * 256 + threadIdx.x;    // NT*128
    int d = (int)(t & 127);
    size_t tok = t >> 7;
    float acc = b_dt[d];
#pragma unroll
    for (int r = 0; r < DR_; ++r)
        acc = fmaf(dbl[tok * DBL_STRIDE + r], W_dt[d * DR_ + r], acc);
    delta[t] = (acc > 20.0f) ? acc : log1pf(__expf(acc));
}

// ===========================================================================
// Scan pass A: per-chunk affine reduction (P = prod a, Q = carry-free state)
// wave = (b, chunk, d-pair); lanes: [dd(2) x n(16)]
// ===========================================================================
__global__ void k_scan_reduce(const float* __restrict__ delta,
                              const float* __restrict__ xm_act,
                              const float* __restrict__ dbl,
                              const float* __restrict__ A_log,
                              float* __restrict__ Pbuf,
                              float* __restrict__ Qbuf) {
    const int lane = threadIdx.x & 31;
    const int gw   = (int)((blockIdx.x * 256u + threadIdx.x) >> 5);
    const int dpair = gw & 63;
    const int c     = (gw >> 6) & 63;
    const int b     = gw >> 12;
    const int d     = dpair * 2 + (lane >> 4);
    const int n     = lane & 15;

    const float Aval = -__expf(A_log[d * DS_ + n]);
    const size_t tok0 = (size_t)b * L_ + (size_t)c * CH_;
    float P = 1.0f, Q = 0.0f;
    for (int s = 0; s < CH_; ++s) {
        size_t tok = tok0 + s;
        float dv = delta[(tok << 7) + d];
        float xv = xm_act[(tok << 7) + d];
        float bv = dbl[tok * DBL_STRIDE + DR_ + n];
        float a  = __expf(dv * Aval);
        float u  = dv * bv * xv;
        P *= a;
        Q = fmaf(Q, a, u);
    }
    size_t idx = ((((size_t)b * NCH_ + c) * DI_ + d) << 4) + n;
    Pbuf[idx] = P; Qbuf[idx] = Q;
}

// ===========================================================================
// Scan pass B: sequential carry over chunks (tiny), emits exclusive h_in
// ===========================================================================
__global__ void k_scan_carry(const float* __restrict__ Pbuf,
                             const float* __restrict__ Qbuf,
                             float* __restrict__ hin) {
    int t = (int)(blockIdx.x * 256u + threadIdx.x);       // B*128*16 = 8192
    int n = t & 15, d = (t >> 4) & 127, b = t >> 11;
    float h = 0.0f;
    for (int c = 0; c < NCH_; ++c) {
        size_t idx = ((((size_t)b * NCH_ + c) * DI_ + d) << 4) + n;
        hin[idx] = h;
        h = fmaf(Pbuf[idx], h, Qbuf[idx]);
    }
}

// ===========================================================================
// Scan pass C: re-run chunks with correct carry-in, emit
//   yact = (y + xm*D) * silu(z)   as f16 (A-matrix of the out-GEMM)
// ===========================================================================
__global__ void k_scan_out(const float* __restrict__ delta,
                           const float* __restrict__ xm_act,
                           const float* __restrict__ dbl,
                           const float* __restrict__ A_log,
                           const float* __restrict__ hin,
                           const float* __restrict__ zbuf,
                           const float* __restrict__ D_param,
                           _Float16* __restrict__ yact) {
    const int lane = threadIdx.x & 31;
    const int gw   = (int)((blockIdx.x * 256u + threadIdx.x) >> 5);
    const int dpair = gw & 63;
    const int c     = (gw >> 6) & 63;
    const int b     = gw >> 12;
    const int d     = dpair * 2 + (lane >> 4);
    const int n     = lane & 15;

    const float Aval = -__expf(A_log[d * DS_ + n]);
    const float Dp   = D_param[d];
    const size_t tok0 = (size_t)b * L_ + (size_t)c * CH_;
    float h = hin[((((size_t)b * NCH_ + c) * DI_ + d) << 4) + n];

    for (int s = 0; s < CH_; ++s) {
        size_t tok = tok0 + s;
        float dv = delta[(tok << 7) + d];
        float xv = xm_act[(tok << 7) + d];
        float bv = dbl[tok * DBL_STRIDE + DR_ + n];
        float cv = dbl[tok * DBL_STRIDE + DR_ + DS_ + n];
        float a  = __expf(dv * Aval);
        h = fmaf(a, h, dv * bv * xv);
        float part = h * cv;
#pragma unroll
        for (int m = 8; m >= 1; m >>= 1) part += __shfl_xor(part, m, 32);
        if (n == 0) {
            float zv = zbuf[(tok << 7) + d];
            float yf = (part + xv * Dp) * siluf_(zv);
            yact[(tok << 7) + d] = (_Float16)yf;
        }
    }
}

// ===========================================================================
// K6: out = yact @ W_out^T (M=NT,K=128,N=64), transpose via LDS, +x -> xr NCHW
// ===========================================================================
__global__ void k_gemm_out(const _Float16* __restrict__ yact,
                           const float* __restrict__ W_out,
                           const float* __restrict__ x,
                           float* __restrict__ xr) {
    __shared__ float ot[C_][132];
    const int tid = threadIdx.x, wv = tid >> 5, lane = tid & 31;
    const int blk = blockIdx.x;
    const int b = blk >> 7, lbase = (blk & 127) * 128;
    const size_t tbase = (size_t)b * L_ + lbase;

    const int nt = wv & 3, mhalf = wv >> 2;
    v16h bf[4];
#pragma unroll
    for (int ks = 0; ks < 4; ++ks)
        bf[ks] = load_b_frag_f32(W_out, 128, nt * 16, ks * 32, 64, lane);

    const int row0 = (lane < 16) ? 0 : 8;
    const int col  = lane & 15;
    for (int mi = 0; mi < 4; ++mi) {
        int mt = mhalf * 4 + mi;
        v8f acc = {};
        const _Float16* arow = yact + ((tbase + mt * 16) << 7);
#pragma unroll
        for (int ks = 0; ks < 4; ++ks)
            acc = wmma_f16(load_a_frag(arow, 128, ks * 32, lane), bf[ks], acc);
#pragma unroll
        for (int r = 0; r < 8; ++r)
            ot[nt * 16 + col][mt * 16 + row0 + r] = acc[r];
    }
    __syncthreads();

    for (int i = tid; i < C_ * 128; i += 256) {
        int c = i >> 7, l = i & 127;
        size_t xi = (size_t)(b * C_ + c) * L_ + lbase + l;
        xr[xi] = ot[c][l] + x[xi];
    }
}

// ===========================================================================
// K7a: skip = xr + dw_h(xr) + dw_w(xr)   (writes skip output in d_out)
// ===========================================================================
__global__ void k_skip(const float* __restrict__ xr,
                       const float* __restrict__ dwh_w,
                       const float* __restrict__ dwh_b,
                       const float* __restrict__ dww_w,
                       const float* __restrict__ dww_b,
                       float* __restrict__ skip) {
    size_t t = (size_t)blockIdx.x * 256 + threadIdx.x;    // B*C*H*W
    int w = (int)(t & 127);
    int h = (int)((t >> 7) & 127);
    int c = (int)((t >> 14) & 63);
    const float* base = xr + (t >> 14 << 14);
    float v = base[h * 128 + w];
    float vh = fmaf(dwh_w[c * 3 + 1], v, dwh_b[c]);
    if (h > 0)   vh = fmaf(dwh_w[c * 3 + 0], base[(h - 1) * 128 + w], vh);
    if (h < 127) vh = fmaf(dwh_w[c * 3 + 2], base[(h + 1) * 128 + w], vh);
    float vw = fmaf(dww_w[c * 3 + 1], v, dww_b[c]);
    if (w > 0)   vw = fmaf(dww_w[c * 3 + 0], base[h * 128 + w - 1], vw);
    if (w < 127) vw = fmaf(dww_w[c * 3 + 2], base[h * 128 + w + 1], vw);
    skip[t] = v + vh + vw;
}

__global__ void k_zero_stats(float* __restrict__ sums) {
    sums[threadIdx.x] = 0.0f;   // 256 floats: [0..127]=sum, [128..255]=sumsq
}

// ===========================================================================
// K7b: pw = skip @ pw_w^T + pw_b  (M=NT,K=64,N=128), NCHW out + BN partials
// ===========================================================================
__global__ void k_gemm_pw(const float* __restrict__ skip,
                          const float* __restrict__ pw_w,
                          const float* __restrict__ pw_b,
                          float* __restrict__ pw,
                          float* __restrict__ sums) {
    __shared__ _Float16 ah[128][72];
    __shared__ float    ot[OC_][132];
    const int tid = threadIdx.x, wv = tid >> 5, lane = tid & 31;
    const int blk = blockIdx.x;
    const int b = blk >> 7, lbase = (blk & 127) * 128;

    for (int i = tid; i < C_ * 128; i += 256) {
        int c = i >> 7, l = i & 127;
        ah[l][c] = (_Float16)skip[(size_t)(b * C_ + c) * L_ + lbase + l];
    }
    __syncthreads();

    const int n0 = wv * 16;
    v16h bf0 = load_b_frag_f32(pw_w, 64, n0, 0,  OC_, lane);
    v16h bf1 = load_b_frag_f32(pw_w, 64, n0, 32, OC_, lane);
    const int row0 = (lane < 16) ? 0 : 8;
    const int col  = lane & 15;
    for (int mt = 0; mt < 8; ++mt) {
        v8f acc = {};
        acc = wmma_f16(load_a_frag(&ah[mt * 16][0], 72, 0,  lane), bf0, acc);
        acc = wmma_f16(load_a_frag(&ah[mt * 16][0], 72, 32, lane), bf1, acc);
#pragma unroll
        for (int r = 0; r < 8; ++r)
            ot[n0 + col][mt * 16 + row0 + r] = acc[r];
    }
    __syncthreads();

    int oc = tid >> 1, half = tid & 1;
    float s = 0.f, sq = 0.f;
    float bias = pw_b[oc];
#pragma unroll 4
    for (int j = 0; j < 64; ++j) {
        int l = half * 64 + j;
        float vv = ot[oc][l] + bias;
        pw[(size_t)(b * OC_ + oc) * L_ + lbase + l] = vv;
        s += vv; sq += vv * vv;
    }
    atomicAdd(&sums[oc], s);
    atomicAdd(&sums[OC_ + oc], sq);
}

// ===========================================================================
// K8: batchnorm (from partial sums) + relu + 2x2 maxpool -> pooled output
// ===========================================================================
__global__ void k_bn_pool(const float* __restrict__ pw,
                          const float* __restrict__ sums,
                          const float* __restrict__ bn_g,
                          const float* __restrict__ bn_b,
                          float* __restrict__ pooled) {
    size_t t = (size_t)blockIdx.x * 256 + threadIdx.x;    // B*OC*64*64
    int wo = (int)(t & 63);
    int ho = (int)((t >> 6) & 63);
    int oc = (int)((t >> 12) & 127);
    int b  = (int)(t >> 19);
    const float invN = 1.0f / (float)(B_ * H_ * W_);
    float m   = sums[oc] * invN;
    float var = sums[OC_ + oc] * invN - m * m;
    float g   = bn_g[oc] * rsqrtf(var + EPS_);
    float bb  = bn_b[oc] - m * g;
    const float* base = pw + ((size_t)(b * OC_ + oc) << 14);
    int h = ho * 2, w = wo * 2;
    float v0 = fmaxf(0.f, fmaf(base[h * 128 + w],           g, bb));
    float v1 = fmaxf(0.f, fmaf(base[h * 128 + w + 1],       g, bb));
    float v2 = fmaxf(0.f, fmaf(base[(h + 1) * 128 + w],     g, bb));
    float v3 = fmaxf(0.f, fmaf(base[(h + 1) * 128 + w + 1], g, bb));
    pooled[t] = fmaxf(fmaxf(v0, v1), fmaxf(v2, v3));
}

// ===========================================================================
// Launcher
// ===========================================================================
extern "C" void kernel_launch(void* const* d_in, const int* in_sizes, int n_in,
                              void* d_out, int out_size, void* d_ws, size_t ws_size,
                              hipStream_t stream) {
    const float* x       = (const float*)d_in[0];
    const float* ln_w    = (const float*)d_in[1];
    const float* ln_b    = (const float*)d_in[2];
    const float* W_in    = (const float*)d_in[3];
    const float* conv_w  = (const float*)d_in[4];
    const float* conv_b  = (const float*)d_in[5];
    const float* W_xproj = (const float*)d_in[6];
    const float* W_dt    = (const float*)d_in[7];
    const float* b_dt    = (const float*)d_in[8];
    const float* A_log   = (const float*)d_in[9];
    const float* D_param = (const float*)d_in[10];
    const float* W_out   = (const float*)d_in[11];
    const float* dwh_w   = (const float*)d_in[12];
    const float* dwh_b   = (const float*)d_in[13];
    const float* dww_w   = (const float*)d_in[14];
    const float* dww_b   = (const float*)d_in[15];
    const float* pw_w    = (const float*)d_in[16];
    const float* pw_b    = (const float*)d_in[17];
    const float* bn_g    = (const float*)d_in[18];
    const float* bn_b    = (const float*)d_in[19];

    char* ws = (char*)d_ws;
    const size_t MB = 1ull << 20;
    // buf0: xm_raw -> (after K3) delta -> (after scan) pw        [32 MB]
    float*    xm_raw  = (float*)(ws + 0);
    float*    deltab  = (float*)(ws + 0);
    float*    pwbuf   = (float*)(ws + 0);
    float*    zbuf    = (float*)(ws + 32 * MB);                 // [32 MB]
    float*    xm_act  = (float*)(ws + 64 * MB);                 // [32 MB]
    float*    xrbuf   = (float*)(ws + 64 * MB);                 // reuse (16 MB)
    _Float16* xm_h    = (_Float16*)(ws + 96 * MB);              // [16 MB]
    _Float16* yact    = (_Float16*)(ws + 96 * MB);              // reuse
    float*    dbl     = (float*)(ws + 112 * MB);                // [12 MB]
    float*    Pbuf    = (float*)(ws + 124 * MB);                // [2 MB]
    float*    Qbuf    = (float*)(ws + 126 * MB);                // [2 MB]
    float*    hin     = (float*)(ws + 128 * MB);                // [2 MB]
    float*    stats   = (float*)(ws + 130 * MB);                // 256 floats

    float* pooled = (float*)d_out;
    float* skip   = (float*)d_out + POOLED_ELEMS;

    dim3 blk(256);
    k_ln_gemm_in <<<512,   blk, 0, stream>>>(x, ln_w, ln_b, W_in, xm_raw, zbuf);
    k_conv_silu  <<<32768, blk, 0, stream>>>(xm_raw, conv_w, conv_b, xm_act, xm_h);
    k_gemm_xproj <<<512,   blk, 0, stream>>>(xm_h, W_xproj, dbl);
    k_delta      <<<32768, blk, 0, stream>>>(dbl, W_dt, b_dt, deltab);
    k_scan_reduce<<<2048,  blk, 0, stream>>>(deltab, xm_act, dbl, A_log, Pbuf, Qbuf);
    k_scan_carry <<<32,    blk, 0, stream>>>(Pbuf, Qbuf, hin);
    k_scan_out   <<<2048,  blk, 0, stream>>>(deltab, xm_act, dbl, A_log, hin,
                                             zbuf, D_param, yact);
    k_gemm_out   <<<512,   blk, 0, stream>>>(yact, W_out, x, xrbuf);
    k_skip       <<<16384, blk, 0, stream>>>(xrbuf, dwh_w, dwh_b, dww_w, dww_b, skip);
    k_zero_stats <<<1,     blk, 0, stream>>>(stats);
    k_gemm_pw    <<<512,   blk, 0, stream>>>(skip, pw_w, pw_b, pwbuf, stats);
    k_bn_pool    <<<8192,  blk, 0, stream>>>(pwbuf, stats, bn_g, bn_b, pooled);
}
